// Phi4Attention_50251117363417
// MI455X (gfx1250) — compile-verified, matrix operand order checked by reading, not run on hardware
//
#include <hip/hip_runtime.h>

// ---------- types ----------
typedef __bf16 v16bf __attribute__((ext_vector_type(16)));
typedef __bf16 v4bf  __attribute__((ext_vector_type(4)));
typedef float  v8f   __attribute__((ext_vector_type(8)));
typedef float  v4f   __attribute__((ext_vector_type(4)));

struct Frag32 { uint4 lo; uint4 hi; };   // 32 bytes == v16bf

static constexpr int S    = 2048;
static constexpr int HID  = 3072;
static constexpr int NH   = 24;
static constexpr int NKV  = 8;
static constexpr int HD   = 128;
static constexpr int ROT  = 96;
static constexpr int OPS  = NH * HD + 2 * NKV * HD;   // 5120
static constexpr float NEGV = -1000000000.0f;

__device__ __forceinline__ unsigned short f32_to_bf16(float f) {
    __bf16 h = (__bf16)f;                 // hardware cvt, RNE
    return __builtin_bit_cast(unsigned short, h);
}

// float4 -> 4x bf16 (packed, 8 bytes) via hardware packed converts
__device__ __forceinline__ uint2 pack4_bf16(float4 v) {
    v4f x = {v.x, v.y, v.z, v.w};
    v4bf y = __builtin_convertvector(x, v4bf);
    return __builtin_bit_cast(uint2, y);
}

// Load one 16x16(x32) bf16 fragment from an LDS row (row stride handled by caller).
// Element e (0..15) of lane (half = lane>>4): k = (e<8 ? 8*half+e : 16+8*half+(e-8)).
__device__ __forceinline__ v16bf ld_frag(const unsigned short* rowp, int half) {
    Frag32 f;
    f.lo = *reinterpret_cast<const uint4*>(rowp + 8 * half);
    f.hi = *reinterpret_cast<const uint4*>(rowp + 16 + 8 * half);
    return __builtin_bit_cast(v16bf, f);
}

// ---------- generic NT GEMM: C(MxN) = A(MxK, f32, k-contig) * B(NxK, bf16, k-contig) ----------
// Block tile 128x128, BK=32, 256 threads (8 waves: 4x2), wave tile 32x64 (2x4 frags).
// Software-pipelined: next tile is register-staged during the WMMA block.
// EP==0: C = acc (f32).  EP==1: C = acc*scale + causal(-1e9)  (scores epilogue).
template <int EP>
__global__ __launch_bounds__(256)
void gemm_bf16_nt(const float* __restrict__ A, const unsigned short* __restrict__ B,
                  float* __restrict__ C,
                  int K, int lda, int ldb, int ldc,
                  long long sAz, long long sBz, long long sCz, int bdiv,
                  float scale)
{
    A += (long long)blockIdx.z * sAz;
    B += (long long)(blockIdx.z / bdiv) * sBz;
    C += (long long)blockIdx.z * sCz;

    __shared__ __align__(16) unsigned short As[128 * 40];  // padded: 80B/row = 20 banks
    __shared__ __align__(16) unsigned short Bs[128 * 40];

    const int tid  = threadIdx.x;
    const int lane = tid & 31;
    const int wid  = tid >> 5;
    const int wm   = wid >> 1;        // 0..3  -> 32-row strip
    const int wn   = wid & 1;         // 0..1  -> 64-col strip
    const int half = lane >> 4;
    const int lr   = lane & 15;

    const int bm = blockIdx.x * 128;
    const int bn = blockIdx.y * 128;

    // ---- per-thread staging addresses (invariant over K-loop) ----
    const float*          aP[4];
    const unsigned short* bP[2];
    int aO[4], bO[2];
#pragma unroll
    for (int i = 0; i < 4; ++i) {
        int c = i * 256 + tid;                 // 0..1023
        int row = c >> 3;
        int kc  = (c & 7) << 2;                // 0,4,...,28
        aP[i] = A + (long long)(bm + row) * lda + kc;
        aO[i] = row * 40 + kc;
    }
#pragma unroll
    for (int i = 0; i < 2; ++i) {
        int c = i * 256 + tid;                 // 0..511
        int row = c >> 2;
        int kc  = (c & 3) << 3;                // 0,8,16,24
        bP[i] = B + (long long)(bn + row) * ldb + kc;
        bO[i] = row * 40 + kc;
    }

    v8f acc[2][4] = {};
    float4 ra[4];
    uint4  rb[2];

    // ---- prologue: stage tile 0 ----
#pragma unroll
    for (int i = 0; i < 4; ++i) ra[i] = *reinterpret_cast<const float4*>(aP[i]);
#pragma unroll
    for (int i = 0; i < 2; ++i) rb[i] = *reinterpret_cast<const uint4*>(bP[i]);
#pragma unroll
    for (int i = 0; i < 4; ++i)
        *reinterpret_cast<uint2*>(&As[aO[i]]) = pack4_bf16(ra[i]);
#pragma unroll
    for (int i = 0; i < 2; ++i)
        *reinterpret_cast<uint4*>(&Bs[bO[i]]) = rb[i];
    __syncthreads();

    int k0 = 0;
    while (true) {
        const bool hasNext = (k0 + 32) < K;
        if (hasNext) {   // issue next-tile global loads; latency hidden behind WMMAs
#pragma unroll
            for (int i = 0; i < 4; ++i)
                ra[i] = *reinterpret_cast<const float4*>(aP[i] + k0 + 32);
#pragma unroll
            for (int i = 0; i < 2; ++i)
                rb[i] = *reinterpret_cast<const uint4*>(bP[i] + k0 + 32);
        }

        v16bf afr[2], bfr[4];
#pragma unroll
        for (int f = 0; f < 2; ++f)
            afr[f] = ld_frag(&As[(wm * 32 + f * 16 + lr) * 40], half);
#pragma unroll
        for (int f = 0; f < 4; ++f)
            bfr[f] = ld_frag(&Bs[(wn * 64 + f * 16 + lr) * 40], half);

#pragma unroll
        for (int i = 0; i < 2; ++i)
#pragma unroll
            for (int j = 0; j < 4; ++j)
                acc[i][j] = __builtin_amdgcn_wmma_f32_16x16x32_bf16(
                    false, afr[i], false, bfr[j], (short)0, acc[i][j], false, false);

        if (!hasNext) break;
        __syncthreads();                       // everyone done reading LDS
#pragma unroll
        for (int i = 0; i < 4; ++i)
            *reinterpret_cast<uint2*>(&As[aO[i]]) = pack4_bf16(ra[i]);
#pragma unroll
        for (int i = 0; i < 2; ++i)
            *reinterpret_cast<uint4*>(&Bs[bO[i]]) = rb[i];
        __syncthreads();
        k0 += 32;
    }

    // ---- epilogue: lane holds col = lr, rows half*8 + e ----
#pragma unroll
    for (int i = 0; i < 2; ++i) {
        int rbase = bm + wm * 32 + i * 16 + half * 8;
#pragma unroll
        for (int j = 0; j < 4; ++j) {
            int col = bn + wn * 64 + j * 16 + lr;
#pragma unroll
            for (int e = 0; e < 8; ++e) {
                int   row = rbase + e;
                float v   = acc[i][j][e];
                if (EP == 1) v = v * scale + ((col <= row) ? 0.0f : NEGV);
                C[(long long)row * ldc + col] = v;
            }
        }
    }
}

// ---------- f32 -> bf16 weight conversion ----------
__global__ void cvt_bf16_kernel(const float* __restrict__ src,
                                unsigned short* __restrict__ dst, long long n) {
    long long i = (long long)blockIdx.x * 256 + threadIdx.x;
    if (i < n) dst[i] = f32_to_bf16(src[i]);
}

// ---------- RoPE + QKV split (q->f32, k->bf16, v->bf16 transposed) ----------
__global__ __launch_bounds__(256)
void rope_split_kernel(const float* __restrict__ qkv,
                       const float* __restrict__ cosv, const float* __restrict__ sinv,
                       float* __restrict__ qf, unsigned short* __restrict__ kbf,
                       unsigned short* __restrict__ vtbf)
{
    long long idx = (long long)blockIdx.x * 256 + threadIdx.x;   // S*OPS total
    int s = (int)(idx / OPS);
    int o = (int)(idx % OPS);
    float x = qkv[idx];
    if (o < NH * HD) {                               // Q
        int h = o >> 7, d = o & 127;
        float val = x;
        if (d < ROT) {
            float c  = cosv[s * ROT + d];
            float sn = sinv[s * ROT + d];
            float oth = (d < ROT / 2) ? -qkv[idx + ROT / 2] : qkv[idx - ROT / 2];
            val = x * c + oth * sn;
        }
        qf[((long long)h * S + s) * HD + d] = val;
    } else if (o < NH * HD + NKV * HD) {             // K
        int oo = o - NH * HD;
        int h = oo >> 7, d = oo & 127;
        float val = x;
        if (d < ROT) {
            float c  = cosv[s * ROT + d];
            float sn = sinv[s * ROT + d];
            float oth = (d < ROT / 2) ? -qkv[idx + ROT / 2] : qkv[idx - ROT / 2];
            val = x * c + oth * sn;
        }
        kbf[((long long)h * S + s) * HD + d] = f32_to_bf16(val);
    } else {                                         // V -> transposed (h, d, s)
        int oo = o - (NH * HD + NKV * HD);
        int h = oo >> 7, d = oo & 127;
        vtbf[((long long)h * HD + d) * S + s] = f32_to_bf16(x);
    }
}

// ---------- rowwise softmax in place (rows of length S) ----------
__global__ __launch_bounds__(256)
void softmax_rows_kernel(float* __restrict__ attn)
{
    float* p  = attn + (long long)blockIdx.x * S;
    int    tid = threadIdx.x;
    __shared__ float red[256];

    float vals[8];
    float m = -3.4e38f;
#pragma unroll
    for (int i = 0; i < 8; ++i) { vals[i] = p[tid + i * 256]; m = fmaxf(m, vals[i]); }
    red[tid] = m; __syncthreads();
    for (int s2 = 128; s2 > 0; s2 >>= 1) {
        if (tid < s2) red[tid] = fmaxf(red[tid], red[tid + s2]);
        __syncthreads();
    }
    m = red[0]; __syncthreads();

    float sum = 0.0f;
#pragma unroll
    for (int i = 0; i < 8; ++i) { vals[i] = expf(vals[i] - m); sum += vals[i]; }
    red[tid] = sum; __syncthreads();
    for (int s2 = 128; s2 > 0; s2 >>= 1) {
        if (tid < s2) red[tid] += red[tid + s2];
        __syncthreads();
    }
    float inv = 1.0f / red[0];
#pragma unroll
    for (int i = 0; i < 8; ++i) p[tid + i * 256] = vals[i] * inv;
}

// ---------- launcher ----------
extern "C" void kernel_launch(void* const* d_in, const int* in_sizes, int n_in,
                              void* d_out, int out_size, void* d_ws, size_t ws_size,
                              hipStream_t stream)
{
    (void)in_sizes; (void)n_in; (void)out_size; (void)ws_size;
    const float* hs    = (const float*)d_in[0];
    const float* cosv  = (const float*)d_in[1];
    const float* sinv  = (const float*)d_in[2];
    // d_in[3] = attention_mask (causal; computed analytically in the scores epilogue)
    const float* w_qkv = (const float*)d_in[4];
    const float* w_o   = (const float*)d_in[5];

    float* out_p  = (float*)d_out;                       // (S, HID)
    float* attn_p = out_p + (long long)S * HID;          // (NH, S, S)

    // workspace layout (256B aligned slabs), total ~151 MB
    char* w = (char*)d_ws;
    auto alloc = [&](size_t bytes) -> char* {
        char* r = w; w += (bytes + 255) & ~size_t(255); return r;
    };
    unsigned short* wqkv_bf = (unsigned short*)alloc((size_t)OPS * HID * 2);
    unsigned short* wo_bf   = (unsigned short*)alloc((size_t)HID * HID * 2);
    float*          qkv_f   = (float*)         alloc((size_t)S * OPS * 4);
    float*          q_f     = (float*)         alloc((size_t)NH * S * HD * 4);
    unsigned short* k_bf    = (unsigned short*)alloc((size_t)NKV * S * HD * 2);
    unsigned short* vT_bf   = (unsigned short*)alloc((size_t)NKV * HD * S * 2);
    float*          ao_f    = (float*)         alloc((size_t)S * HID * 4);

    const float scaling = 0.08838834764831845f;          // 128^-0.5

    // 1) weight conversions
    {
        long long n1 = (long long)OPS * HID;
        cvt_bf16_kernel<<<(int)((n1 + 255) / 256), 256, 0, stream>>>(w_qkv, wqkv_bf, n1);
        long long n2 = (long long)HID * HID;
        cvt_bf16_kernel<<<(int)((n2 + 255) / 256), 256, 0, stream>>>(w_o, wo_bf, n2);
    }
    // 2) QKV GEMM: (2048x3072)x(3072x5120) -> qkv_f
    gemm_bf16_nt<0><<<dim3(S / 128, OPS / 128, 1), 256, 0, stream>>>(
        hs, wqkv_bf, qkv_f, HID, HID, HID, OPS, 0, 0, 0, 1, 1.0f);
    // 3) RoPE + split
    rope_split_kernel<<<(int)(((long long)S * OPS) / 256), 256, 0, stream>>>(
        qkv_f, cosv, sinv, q_f, k_bf, vT_bf);
    // 4) scores per head: Q(2048x128) x K^T -> attn (scaled + causal mask)
    gemm_bf16_nt<1><<<dim3(S / 128, S / 128, NH), 256, 0, stream>>>(
        q_f, k_bf, attn_p, HD, HD, HD, S,
        (long long)S * HD, (long long)S * HD, (long long)S * S, NH / NKV, scaling);
    // 5) softmax rows in place (final attn_weights output)
    softmax_rows_kernel<<<NH * S, 256, 0, stream>>>(attn_p);
    // 6) P @ V: (2048x2048) x (2048x128) using transposed V -> attn_out columns
    gemm_bf16_nt<0><<<dim3(S / 128, HD / 128, NH), 256, 0, stream>>>(
        attn_p, vT_bf, ao_f, S, S, S, HID,
        (long long)S * S, (long long)HD * S, (long long)HD, NH / NKV, 1.0f);
    // 7) output projection: (2048x3072) x (3072x3072) -> out
    gemm_bf16_nt<0><<<dim3(S / 128, HID / 128, 1), 256, 0, stream>>>(
        ao_f, wo_bf, out_p, HID, HID, HID, HID, 0, 0, 0, 1, 1.0f);
}
